// GRU_48627619726124
// MI455X (gfx1250) — matrix operand order repaired and checked by reading, not compiled, not run on hardware
//
#include <hip/hip_runtime.h>

typedef __attribute__((ext_vector_type(16))) _Float16 v16h;
typedef __attribute__((ext_vector_type(8)))  _Float16 v8h;
typedef __attribute__((ext_vector_type(2)))  _Float16 v2h;
typedef __attribute__((ext_vector_type(8)))  float    v8f;
typedef __attribute__((ext_vector_type(2)))  float    v2f;

#define GRU_B 4096
#define GRU_T 128
#define GRU_I 16
#define GRU_H 64

#define NCOLS   256        // 4 gate groups of H: [r | z | gi_n | hn]
#define KDIM    96         // 64 (h) + 16 (x) + 1 (bias one-hot) + 15 pad
#define KTILES  3
#define ROWS    16         // batch rows per workgroup (one WMMA M-tile)
#define WAVES   4          // wave w owns N-tiles {w, 4+w, 8+w, 12+w}
#define ASTRIDE 104        // padded f16 stride of A staging rows (208B = 52 banks)

__device__ __forceinline__ float sigmoid_f(float v) {
  return __builtin_amdgcn_rcpf(1.0f + __expf(-v));
}
__device__ __forceinline__ float tanh_f(float v) {
  return 1.0f - 2.0f * __builtin_amdgcn_rcpf(__expf(2.0f * v) + 1.0f);
}

__global__ __launch_bounds__(WAVES * 32) void gru_wmma_kernel(
    const float* __restrict__ x,    const float* __restrict__ w_ih,
    const float* __restrict__ w_hh, const float* __restrict__ b_ih,
    const float* __restrict__ b_hh, const float* __restrict__ fc_w,
    const float* __restrict__ fc_b, float* __restrict__ out)
{
  // B^T: [n=0..255][k=0..95] f16 (49152 B); read once into registers, then
  // reused as f32 h buffer for the final head.
  __shared__ __align__(16) _Float16 sBT[NCOLS * KDIM];
  // shared A staging: 16 rows x 104 f16. cols: 0..63 h, 64..79 x_t,
  // 80 = 1.0 (bias selector), 81..95 = 0
  __shared__ __align__(16) _Float16 sStage[ROWS * ASTRIDE];

  const int tid  = threadIdx.x;
  const int lane = tid & 31;
  const int wave = tid >> 5;
  const int b0   = blockIdx.x * ROWS;

  // ---- build fused gate matrix B^T in LDS (once, then hoisted to VGPRs) ----
  for (int n = tid; n < NCOLS; n += WAVES * 32) {
    for (int k = 0; k < KDIM; ++k) {
      float v = 0.0f;
      if (n < 128) {                       // r (0..63) and z (64..127): full sum
        if (k < 64)       v = w_hh[n * 64 + k];
        else if (k < 80)  v = w_ih[n * 16 + (k - 64)];
        else if (k == 80) v = b_ih[n] + b_hh[n];
      } else if (n < 192) {                // gi_n: x-path of n gate only
        if (k >= 64 && k < 80) v = w_ih[n * 16 + (k - 64)];
        else if (k == 80)      v = b_ih[n];
      } else {                             // hn: h-path of n gate only
        const int s = n - 64;              // w_hh rows 128..191
        if (k < 64)       v = w_hh[s * 64 + k];
        else if (k == 80) v = b_hh[s];
      }
      sBT[n * KDIM + k] = (_Float16)v;
    }
  }
  // ---- init A staging ----
  for (int i = tid; i < ROWS * ASTRIDE; i += WAVES * 32)
    sStage[i] = (_Float16)0.0f;
  if (tid < ROWS)
    sStage[tid * ASTRIDE + 80] = (_Float16)1.0f;   // bias selector column
  __syncthreads();

  // ---- hoist this wave's 12 B fragments into registers (stationary) ----
  const int nlo = lane & 15;
  const int khi = (lane >> 4) << 4;        // B layout: lanes 0-15 K 0-15, 16-31 K 16-31
  v16h bfr[KTILES][4];
#pragma unroll
  for (int kt = 0; kt < KTILES; ++kt) {
#pragma unroll
    for (int g = 0; g < 4; ++g) {
      const int n = (g * 4 + wave) * 16 + nlo;   // group g, column slice w
      const v8h lo = *(const v8h*)&sBT[n * KDIM + kt * 32 + khi];
      const v8h hi = *(const v8h*)&sBT[n * KDIM + kt * 32 + khi + 8];
      bfr[kt][g] = __builtin_shufflevector(lo, hi,
          0,1,2,3,4,5,6,7,8,9,10,11,12,13,14,15);
    }
  }

  const int arow  = lane & 15;
  const int ahalf = lane >> 4;             // A layout half select

  float hreg[8];                           // h columns wave*16+nlo, D layout rows
#pragma unroll
  for (int j = 0; j < 8; ++j) hreg[j] = 0.0f;

  // cooperative x staging: thread covers row = tid>>3, 2 cols at (tid&7)*2
  const int xrow  = tid >> 3;
  const int xpair = (tid & 7) * 2;
  const float* xbase = x + (size_t)(b0 + xrow) * GRU_T * GRU_I + xpair;
  const int hcol = wave * 16 + nlo;

  for (int t = 0; t < GRU_T; ++t) {
    // stage x_t (f32 -> f16) into cols 64..79 (coalesced across the WG)
    {
      const v2f f = *(const v2f*)(xbase + t * GRU_I);
      if (t + 1 < GRU_T) __builtin_prefetch(xbase + (t + 1) * GRU_I, 0, 1);
      v2h p; p[0] = (_Float16)f[0]; p[1] = (_Float16)f[1];
      *(v2h*)&sStage[xrow * ASTRIDE + 64 + xpair] = p;
    }
    __syncthreads();                       // (1) h_t and x_t staged & visible

    // A fragments: two contiguous b128 loads, k = 16*half + s (same order as B)
    v16h a[KTILES];
#pragma unroll
    for (int kt = 0; kt < KTILES; ++kt) {
      const v8h a0 = *(const v8h*)&sStage[arow * ASTRIDE + kt * 32 + ahalf * 16];
      const v8h a1 = *(const v8h*)&sStage[arow * ASTRIDE + kt * 32 + ahalf * 16 + 8];
      a[kt] = __builtin_shufflevector(a0, a1,
          0,1,2,3,4,5,6,7,8,9,10,11,12,13,14,15);
    }
    __syncthreads();                       // (2) staging consumed by all waves

    v8f acc[4];
#pragma unroll
    for (int g = 0; g < 4; ++g) acc[g] = (v8f){0,0,0,0,0,0,0,0};
#pragma unroll
    for (int kt = 0; kt < KTILES; ++kt) {
#pragma unroll
      for (int g = 0; g < 4; ++g) {
        acc[g] = __builtin_amdgcn_wmma_f32_16x16x32_f16(
            false, a[kt], false, bfr[kt][g], (short)0, acc[g], false, false);
      }
    }

    // gate math: r/z/gi_n/hn line up elementwise across the 4 acc tiles
#pragma unroll
    for (int j = 0; j < 8; ++j) {
      const float r  = sigmoid_f(acc[0][j]);
      const float z  = sigmoid_f(acc[1][j]);
      const float nn = tanh_f(acc[2][j] + r * acc[3][j]);
      const float hnew = (1.0f - z) * nn + z * hreg[j];
      hreg[j] = hnew;
      sStage[(j + ahalf * 8) * ASTRIDE + hcol] = (_Float16)hnew;  // next A
    }
  }

  // ---- final head: pred[b] = h . fc_w + fc_b (f32 h via reused sBT space) ----
  float* hd = reinterpret_cast<float*>(sBT);   // 16 x 64 f32 (4 KB)
#pragma unroll
  for (int j = 0; j < 8; ++j)
    hd[(j + ahalf * 8) * GRU_H + hcol] = hreg[j];
  __syncthreads();
  if (tid < ROWS) {
    float s = fc_b[0];
    const float* hr = hd + tid * GRU_H;
#pragma unroll 16
    for (int k = 0; k < GRU_H; ++k) s += hr[k] * fc_w[k];
    out[b0 + tid] = s;
  }
}

extern "C" void kernel_launch(void* const* d_in, const int* in_sizes, int n_in,
                              void* d_out, int out_size, void* d_ws, size_t ws_size,
                              hipStream_t stream) {
  (void)in_sizes; (void)n_in; (void)out_size; (void)d_ws; (void)ws_size;
  const float* x    = (const float*)d_in[0];
  const float* w_ih = (const float*)d_in[1];
  const float* w_hh = (const float*)d_in[2];
  const float* b_ih = (const float*)d_in[3];
  const float* b_hh = (const float*)d_in[4];
  const float* fc_w = (const float*)d_in[5];
  const float* fc_b = (const float*)d_in[6];
  float* out = (float*)d_out;

  dim3 grid(GRU_B / ROWS);            // 256 workgroups (one M-tile each)
  dim3 block(WAVES * 32);             // 128 threads = 4 waves
  hipLaunchKernelGGL(gru_wmma_kernel, grid, block, 0, stream,
                     x, w_ih, w_hh, b_ih, b_hh, fc_w, fc_b, out);
}
